// NeighborMLPConvLayerLinear_15350213116606
// MI455X (gfx1250) — matrix operand order repaired and checked by reading, not compiled, not run on hardware
//
#include <hip/hip_runtime.h>
#include <hip/hip_bf16.h>

typedef __attribute__((ext_vector_type(16))) __bf16 v16bf;
typedef __attribute__((ext_vector_type(8)))  float  v8f;

#define C_IN  32
#define HID   64
#define C_OUT 32
#define WAVES_PER_BLOCK 8

__device__ __forceinline__ v8f wmma_bf16(v16bf a, v16bf b, v8f c) {
  // D = A(16x32 bf16) * B(32x16 bf16) + C(16x16 f32)
  return __builtin_amdgcn_wmma_f32_16x16x32_bf16(
      /*neg_a=*/false, a, /*neg_b=*/false, b,
      /*c_mod=*/(short)0, c, /*reuse_a=*/false, /*reuse_b=*/false);
}

// GELU via gfx1250 hardware tanh (V_TANH_F32): branch-free, ~6 VALU ops,
// max err ~3e-4 (below the bf16 noise already in the matmul path).
__device__ __forceinline__ float gelu_fast(float x) {
  const float k0 = 0.7978845608028654f;  // sqrt(2/pi)
  const float k1 = 0.044715f;
  float u = k0 * __builtin_fmaf(k1 * x * x, x, x);
  float t;
#if __has_builtin(__builtin_amdgcn_tanhf)
  t = __builtin_amdgcn_tanhf(u);
#else
  // TRANS-op result hazard: 1 independent op needed before use -> pad with v_nop.
  asm volatile("v_tanh_f32 %0, %1\n\tv_nop\n\tv_nop" : "=v"(t) : "v"(u));
#endif
  return 0.5f * x * (1.0f + t);
}

// Load a 32x16 bf16 B-fragment from row-major f32 W[k][n] (leading dim ld).
// Layout: element s, lanes 0-15 hold K 0-7 / 16-23, lanes 16-31 hold K 8-15 / 24-31 (N=lane).
__device__ __forceinline__ v16bf load_bfrag(const float* __restrict__ W, int ld,
                                            int k0, int n0, int lane) {
  const int n  = n0 + (lane & 15);
  const int kb = k0 + ((lane & 16) >> 1);   // +8 for upper-half lanes
  v16bf b;
#pragma unroll
  for (int v = 0; v < 8; ++v) {
    b[2 * v]     = (__bf16)W[(kb + 2 * v + (v >= 4 ? 8 : 0))     * ld + n];
    b[2 * v + 1] = (__bf16)W[(kb + 2 * v + 1 + (v >= 4 ? 8 : 0)) * ld + n];
  }
  return b;
}

// Build one lane's share of a 16x32 bf16 A-fragment from a 32-float row.
// off = 0 (lanes 0-15: K 0-7 / 16-23) or 8 (lanes 16-31: K 8-15 / 24-31).
__device__ __forceinline__ v16bf make_afrag(const float* __restrict__ row, int off) {
  v16bf a;
#pragma unroll
  for (int j = 0; j < 8; ++j) a[j]     = (__bf16)row[off + j];
#pragma unroll
  for (int j = 0; j < 8; ++j) a[8 + j] = (__bf16)row[off + 16 + j];
  return a;
}

__global__ __launch_bounds__(256) void neigh_mlp_conv_kernel(
    const float* __restrict__ x_in,     // [N, C_IN]
    const float* __restrict__ x_out,    // [M, C_IN]
    const float* __restrict__ in_feat,  // [N, C_OUT]
    const int*   __restrict__ nidx,     // [E]
    const int*   __restrict__ rsplit,   // [M+1]
    const float* __restrict__ W1,       // [2*C_IN, HID]
    const float* __restrict__ b1,       // [HID]
    const float* __restrict__ W2,       // [HID, C_OUT]
    const float* __restrict__ b2,       // [C_OUT]
    float* __restrict__ out,            // [M, C_OUT]
    int Mpts) {
  __shared__ float lds_h[WAVES_PER_BLOCK][16 * HID];  // 4 KB per wave

  const int lane  = threadIdx.x & 31;
  const int wave  = threadIdx.x >> 5;
  const int gwave = blockIdx.x * WAVES_PER_BLOCK + wave;
  const int nwave = gridDim.x * WAVES_PER_BLOCK;
  const int off   = (lane & 16) >> 1;  // 0 or 8 (A-fragment K sub-offset / C row base)

  // ---- Weight fragments in registers, bf16, loaded once per wave ----
  v16bf w1f[2][4];  // [k-chunk][n-tile] for the 64x64 W1
  v16bf w2f[2][2];  // [k-chunk][n-tile] for the 64x32 W2
#pragma unroll
  for (int kc = 0; kc < 2; ++kc) {
#pragma unroll
    for (int nt = 0; nt < 4; ++nt)
      w1f[kc][nt] = load_bfrag(W1, HID, kc * 32, nt * 16, lane);
#pragma unroll
    for (int nt = 0; nt < 2; ++nt)
      w2f[kc][nt] = load_bfrag(W2, C_OUT, kc * 32, nt * 16, lane);
  }
  float b1v[4], b2v[2];
#pragma unroll
  for (int nt = 0; nt < 4; ++nt) b1v[nt] = b1[nt * 16 + (lane & 15)];
#pragma unroll
  for (int nt = 0; nt < 2; ++nt) b2v[nt] = b2[nt * 16 + (lane & 15)];

  float* hb = lds_h[wave];

  // ---- One output point per wave iteration (segment reduce stays in-wave) ----
  for (int m = gwave; m < Mpts; m += nwave) {
    const int e0  = rsplit[m];
    const int cnt = rsplit[m + 1] - e0;  // reference: always 32
    int myidx = (lane < cnt) ? nidx[e0 + lane] : 0;

    // Self-feature half of the concat row: identical for all 32 edges of m.
    const v16bf a_self = make_afrag(x_out + (size_t)m * C_IN, off);

    float osum0 = 0.f, osum1 = 0.f;

#pragma unroll
    for (int t = 0; t < 2; ++t) {  // two 16-edge M-tiles
      const int er     = t * 16 + (lane & 15);
      const int gi_row = __shfl(myidx, er, 32);

      v16bf a_rep;
      if (er < cnt) {
        a_rep = make_afrag(x_in + (size_t)gi_row * C_IN, off);
      } else {
#pragma unroll
        for (int j = 0; j < 16; ++j) a_rep[j] = (__bf16)0.0f;
      }

      // ---- GEMM1: h[16x64] = [a_rep | a_self] @ W1 ----
      v8f h[4] = {};
#pragma unroll
      for (int nt = 0; nt < 4; ++nt) {
        h[nt] = wmma_bf16(a_rep,  w1f[0][nt], h[nt]);
        h[nt] = wmma_bf16(a_self, w1f[1][nt], h[nt]);
      }

      // Bias + GELU (hw tanh), spill to LDS row-major to re-pack C->A layout.
#pragma unroll
      for (int nt = 0; nt < 4; ++nt) {
        const int col = nt * 16 + (lane & 15);
#pragma unroll
        for (int p = 0; p < 8; ++p) {
          const float g = gelu_fast(h[nt][p] + b1v[nt]);
          hb[(off + p) * HID + col] = g;  // row = p + (lane>=16 ? 8 : 0)
        }
      }

      // Reload as A fragments (same-wave DS ordering covers the RAW hazard).
      const float* hrow = hb + (lane & 15) * HID;
      const v16bf g0 = make_afrag(hrow, off);        // K = 0..31
      const v16bf g1 = make_afrag(hrow + 32, off);   // K = 32..63

      // ---- GEMM2: o[16x32] = gelu(h) @ W2 ----
      v8f o[2] = {};
#pragma unroll
      for (int nt = 0; nt < 2; ++nt) {
        o[nt] = wmma_bf16(g0, w2f[0][nt], o[nt]);
        o[nt] = wmma_bf16(g1, w2f[1][nt], o[nt]);
      }

      // ---- (o + b2) * in_features[nidx]  and in-register row-sum ----
#pragma unroll
      for (int nt = 0; nt < 2; ++nt) {
        const int col = nt * 16 + (lane & 15);
        float s = 0.f;
#pragma unroll
        for (int p = 0; p < 8; ++p) {
          const int e  = t * 16 + off + p;       // edge row for this C element
          const int gi = __shfl(myidx, e, 32);   // uniform participation
          float feat = 0.f;
          if (e < cnt) feat = in_feat[(size_t)gi * C_OUT + col];
          s += (o[nt][p] + b2v[nt]) * feat;
        }
        if (nt == 0) osum0 += s; else osum1 += s;
      }
    }  // t

    // Merge the two row-halves (lane n holds rows 0-7 col n, lane n+16 rows 8-15).
    osum0 += __shfl_xor(osum0, 16, 32);
    osum1 += __shfl_xor(osum1, 16, 32);

    const float inv = 1.0f / (float)(cnt > 0 ? cnt : 1);
    if (lane < 16) {
      out[(size_t)m * C_OUT + lane]      = osum0 * inv;
      out[(size_t)m * C_OUT + 16 + lane] = osum1 * inv;
    }
  }
}

extern "C" void kernel_launch(void* const* d_in, const int* in_sizes, int n_in,
                              void* d_out, int out_size, void* d_ws, size_t ws_size,
                              hipStream_t stream) {
  const float* x_in    = (const float*)d_in[0];
  const float* x_out   = (const float*)d_in[1];
  const float* in_feat = (const float*)d_in[2];
  const int*   nidx    = (const int*)d_in[3];
  const int*   rsplit  = (const int*)d_in[4];
  const float* W1      = (const float*)d_in[5];
  const float* b1      = (const float*)d_in[6];
  const float* W2      = (const float*)d_in[7];
  const float* b2      = (const float*)d_in[8];
  float* out = (float*)d_out;

  const int Mpts = in_sizes[1] / C_IN;  // x_out is [M, C_IN]

  int blocks = 1024;  // 8192 waves; ~6 points per wave amortizes weight-fragment loads
  const int need = (Mpts + WAVES_PER_BLOCK - 1) / WAVES_PER_BLOCK;
  if (blocks > need) blocks = need;
  if (blocks < 1) blocks = 1;

  neigh_mlp_conv_kernel<<<blocks, 256, 0, stream>>>(
      x_in, x_out, in_feat, nidx, rsplit, W1, b1, W2, b2, out, Mpts);
}
